// VectorQuantizerEMA_18004502905465
// MI455X (gfx1250) — compile-verified, matrix operand order checked by reading, not run on hardware
//
#include <hip/hip_runtime.h>
#include <hip/hip_bf16.h>
#include <math.h>

#define Bk 16
#define Dk 128
#define Tk 4096
#define Kk 1024
#define Nk (Bk * Tk)                 // 65536
#define QOFF 1ull                    // quantized starts at d_out[1]
#define QNUM 8388608ull              // 16*128*4096
#define POFF (QOFF + QNUM)           // perplexity at 8388609
#define IOFF (POFF + 1ull)           // idx at 8388610

typedef __attribute__((ext_vector_type(16))) __bf16 v16bf;
typedef __attribute__((ext_vector_type(8)))  float  v8f;

union Frag {
    v16bf v;
    unsigned short u[16];
    uint4 q[2];
};

__device__ __forceinline__ unsigned short f2bf(float f) {
    union { float f; unsigned int u; } c; c.f = f;
    unsigned int u = c.u;
    unsigned int r = u + 0x7FFFu + ((u >> 16) & 1u);   // round-to-nearest-even
    return (unsigned short)(r >> 16);
}

// ---------------------------------------------------------------------------
// Kernel 1: codebook -> bf16, per-code squared norms, zero counts/loss.
// grid = 1024 blocks (one per code), 128 threads (one per d).
// ---------------------------------------------------------------------------
__global__ __launch_bounds__(128)
void vq_prep(const float* __restrict__ cb, unsigned short* __restrict__ cb_bf,
             float* __restrict__ cb_norm, float* __restrict__ counts,
             float* __restrict__ lossAcc) {
    __shared__ float red[128];
    const int k = blockIdx.x;
    const int d = threadIdx.x;
    float v = cb[k * Dk + d];
    cb_bf[k * Dk + d] = f2bf(v);
    red[d] = v * v;
    __syncthreads();
    for (int s = 64; s > 0; s >>= 1) {
        if (d < s) red[d] += red[d + s];
        __syncthreads();
    }
    if (d == 0) {
        cb_norm[k] = red[0];
        counts[k] = 0.0f;
        if (k == 0) *lossAcc = 0.0f;
    }
}

// ---------------------------------------------------------------------------
// Kernel 2: block = 128 threads (4 waves), 128 flat rows (one b, 128 t's).
// Each wave owns TWO 16-row A tiles and reuses every B fragment for two
// WMMAs (8 wmma : 8 b128-loads per code block). Prefetch next code block.
// grid = 512 blocks.
// ---------------------------------------------------------------------------
__global__ __launch_bounds__(128)
void vq_main(const float* __restrict__ x, const float* __restrict__ cb,
             const unsigned short* __restrict__ cbh,
             const float* __restrict__ cbn, float* __restrict__ counts,
             float* __restrict__ lossAcc, float* __restrict__ out) {
    __shared__ unsigned short xl[Dk * 128];   // staged tile, [d][t] bf16
    __shared__ int idxl[128];                 // winning code per local row

    const int tid = threadIdx.x;
    const int blk = blockIdx.x;               // 0..511
    const int b   = blk >> 5;                 // 32 tiles per batch element
    const int t0  = (blk & 31) << 7;          // tile start t
    const size_t xbase = (size_t)b * Dk * Tk + (size_t)t0;

    // ---- stage input tile: 128 d x 128 t, coalesced float4 along t ----
    for (int i = tid; i < Dk * 128 / 4; i += 128) {
        int d  = i >> 5;                      // 32 float4 per d-row
        int t4 = (i & 31) << 2;
        const float4 f = *(const float4*)(x + xbase + (size_t)d * Tk + t4);
        unsigned short* p = &xl[d * 128 + t4];
        p[0] = f2bf(f.x); p[1] = f2bf(f.y); p[2] = f2bf(f.z); p[3] = f2bf(f.w);
    }
    __syncthreads();

    const int lane = tid & 31;
    const int wv   = tid >> 5;                // wave 0..3
    const int hi   = lane >> 4;               // half-wave (K offset +8)
    const int ln   = lane & 15;               // M / N within fragment
    const int n0loc = wv * 32 + ln;           // tile-0 row (local t)
    const int n1loc = wv * 32 + 16 + ln;      // tile-1 row (local t)

    // ---- A fragments for both tiles (16x32 bf16 x 4 chunks each) ----
    Frag a0[4], a1[4];
    #pragma unroll
    for (int ch = 0; ch < 4; ++ch) {
        const int kb = ch * 32 + hi * 8;
        #pragma unroll
        for (int j = 0; j < 8; ++j) {
            const int K = kb + ((j < 4) ? 2 * j : 16 + 2 * (j - 4));
            a0[ch].u[2 * j]     = xl[K * 128 + n0loc];
            a0[ch].u[2 * j + 1] = xl[(K + 1) * 128 + n0loc];
            a1[ch].u[2 * j]     = xl[K * 128 + n1loc];
            a1[ch].u[2 * j + 1] = xl[(K + 1) * 128 + n1loc];
        }
    }

    float bestD0[8], bestD1[8];
    int   bestI0[8], bestI1[8];
    #pragma unroll
    for (int r = 0; r < 8; ++r) {
        bestD0[r] = 3.4e38f; bestI0[r] = 0;
        bestD1[r] = 3.4e38f; bestI1[r] = 0;
    }

    // ---- sweep 64 blocks of 16 codes ----
    for (int cblk = 0; cblk < Kk / 16; ++cblk) {
        const int c = cblk * 16 + ln;         // this lane's code column
        const unsigned short* crow = cbh + c * Dk;

        // warm L2->WGP path for next code block's row (256B = 2 prefetches)
        if (cblk < Kk / 16 - 1) {
            __builtin_prefetch(crow + 16 * Dk, 0, 3);
            __builtin_prefetch(crow + 16 * Dk + 64, 0, 3);
        }

        // all four chunk B fragments up front (one clause of 8 b128 loads)
        Frag bfr[4];
        #pragma unroll
        for (int ch = 0; ch < 4; ++ch) {
            const uint4* p = (const uint4*)(crow + ch * 32 + hi * 8);
            bfr[ch].q[0] = p[0];              // K = base+0..7   (pairs)
            bfr[ch].q[1] = p[2];              // K = base+16..23 (pairs)
        }

        v8f acc0 = {}, acc1 = {};
        #pragma unroll
        for (int ch = 0; ch < 4; ++ch) {
            acc0 = __builtin_amdgcn_wmma_f32_16x16x32_bf16(
                false, a0[ch].v, false, bfr[ch].v, (short)0, acc0, false, false);
            acc1 = __builtin_amdgcn_wmma_f32_16x16x32_bf16(
                false, a1[ch].v, false, bfr[ch].v, (short)0, acc1, false, false);
        }

        const float en = cbn[c];
        #pragma unroll
        for (int r = 0; r < 8; ++r) {
            float d0 = fmaf(-2.0f, acc0[r], en);   // ||e||^2 - 2 x.e
            float d1 = fmaf(-2.0f, acc1[r], en);
            if (d0 < bestD0[r]) { bestD0[r] = d0; bestI0[r] = c; }
            if (d1 < bestD1[r]) { bestD1[r] = d1; bestI1[r] = c; }
        }
    }

    // ---- argmin across the 16 lanes of each half (codes mod 16) ----
    #pragma unroll
    for (int m = 1; m < 16; m <<= 1) {
        #pragma unroll
        for (int r = 0; r < 8; ++r) {
            float od = __shfl_xor(bestD0[r], m, 32);
            int   oi = __shfl_xor(bestI0[r], m, 32);
            if (od < bestD0[r] || (od == bestD0[r] && oi < bestI0[r])) {
                bestD0[r] = od; bestI0[r] = oi;
            }
            od = __shfl_xor(bestD1[r], m, 32);
            oi = __shfl_xor(bestI1[r], m, 32);
            if (od < bestD1[r] || (od == bestD1[r] && oi < bestI1[r])) {
                bestD1[r] = od; bestI1[r] = oi;
            }
        }
    }
    if (ln == 0) {
        #pragma unroll
        for (int r = 0; r < 8; ++r) {
            idxl[wv * 32 + hi * 8 + r]      = bestI0[r];  // C row = r (+8 hi)
            idxl[wv * 32 + 16 + hi * 8 + r] = bestI1[r];
        }
    }
    __syncthreads();

    // ---- outputs: indices, histogram, gather-quantize, loss ----
    const int n0 = blk * 128;
    {
        const int ci = idxl[tid];
        atomicAdd(&counts[ci], 1.0f);
        out[IOFF + (size_t)(n0 + tid)] = (float)ci;
    }
    float lsum = 0.0f;
    for (int e = tid; e < Dk * 128; e += 128) {
        const int d  = e >> 7;
        const int tl = e & 127;
        const float q  = cb[idxl[tl] * Dk + d];      // exact f32 gather
        const size_t off = xbase + (size_t)d * Tk + tl;
        const float xv = x[off];
        out[QOFF + off] = q;                         // quantized_st == quantized (fwd)
        const float df = q - xv;
        lsum += df * df;
    }
    #pragma unroll
    for (int m = 1; m < 32; m <<= 1) lsum += __shfl_xor(lsum, m, 32);
    if (lane == 0) atomicAdd(lossAcc, lsum);
}

// ---------------------------------------------------------------------------
// Kernel 3: perplexity + loss scalars. 1 block, 256 threads.
// ---------------------------------------------------------------------------
__global__ __launch_bounds__(256)
void vq_final(const float* __restrict__ counts, const float* __restrict__ lossAcc,
              float* __restrict__ out) {
    __shared__ float red[256];
    const int t = threadIdx.x;
    float s = 0.0f;
    for (int i = t; i < Kk; i += 256) {
        const float p = counts[i] * (1.0f / (float)Nk);
        s += p * logf(p + 1e-10f);
    }
    red[t] = s;
    __syncthreads();
    for (int st = 128; st > 0; st >>= 1) {
        if (t < st) red[t] += red[t + st];
        __syncthreads();
    }
    if (t == 0) {
        out[POFF] = expf(-red[0]);
        out[0]    = 0.25f * (*lossAcc) / (float)QNUM;   // commitment * mean sq err
    }
}

extern "C" void kernel_launch(void* const* d_in, const int* in_sizes, int n_in,
                              void* d_out, int out_size, void* d_ws, size_t ws_size,
                              hipStream_t stream) {
    const float* x  = (const float*)d_in[0];   // [16, 128, 4096] f32
    const float* cb = (const float*)d_in[1];   // [1024, 128] f32
    float* out = (float*)d_out;

    char* ws = (char*)d_ws;
    unsigned short* cb_bf = (unsigned short*)ws;                 // 262144 B
    float* cb_norm = (float*)(ws + 262144);                      // 4096 B
    float* counts  = (float*)(ws + 262144 + 4096);               // 4096 B
    float* lossAcc = (float*)(ws + 262144 + 8192);               // 4 B

    vq_prep <<<Kk, 128, 0, stream>>>(cb, cb_bf, cb_norm, counts, lossAcc);
    vq_main <<<Nk / 128, 128, 0, stream>>>(x, cb, cb_bf, cb_norm, counts, lossAcc, out);
    vq_final<<<1, 256, 0, stream>>>(counts, lossAcc, out);
}
